// ContinuousFusion_5050881540166
// MI455X (gfx1250) — compile-verified, matrix operand order checked by reading, not run on hardware
//
#include <hip/hip_runtime.h>
#include <hip/hip_fp16.h>

// ---------------------------------------------------------------------------
// ContinuousFusion for MI455X (gfx1250, wave32, WMMA).
// Pipeline: wconv (f16 weights, K padded to 992) -> argmax (LDS-tiled)
//        -> feature gather/bilinear (f16 X0) -> fused WMMA MLP (all 5 layers
//           + 8 z-channels in one kernel; W2/3/4 async-staged into LDS and
//           overlapped with the layer-1 WMMA sweep; intermediates LDS-only).
// ---------------------------------------------------------------------------

typedef __attribute__((ext_vector_type(16))) _Float16 v16h;
typedef __attribute__((ext_vector_type(8)))  _Float16 v8h;
typedef __attribute__((ext_vector_type(8)))  float    v8f;

#define CHA   8
#define COLS  128
#define ROWS  128
#define NCELL (COLS*ROWS)   // 16384 grid cells per batch
#define NPTS  4096
#define K1    963
#define K1P   992           // 963 padded to 31*32 for 16x16x32 WMMA
#define HID   128
#define WPAD  136           // padded LDS row stride (halfs): 272B -> bank+4/row

union FragU { v16h v; v8h h[2]; };

// ---------------------------------------------------------------------------
// Kernel 1: convert weights to f16. W1 (128x963)->W1h (128x992, zero pad),
// W2/3/4 (128x128 each) -> W234h, and extract wz = W1[:,962] as f32.
// ---------------------------------------------------------------------------
__global__ void wconv_kernel(const float* __restrict__ W1,
                             const float* __restrict__ W2,
                             const float* __restrict__ W3,
                             const float* __restrict__ W4,
                             _Float16* __restrict__ W1h,
                             _Float16* __restrict__ W234h,
                             float* __restrict__ wz) {
  int tid = blockIdx.x * 256 + threadIdx.x;
  if (tid < 128 * K1P) {
    int n = tid / K1P, k = tid % K1P;
    float v = (k < K1) ? W1[n * K1 + k] : 0.0f;
    W1h[tid] = (_Float16)v;
    if (k == K1 - 1) wz[n] = v;
  }
  int t2 = tid - 128 * K1P;
  if (t2 >= 0 && t2 < 3 * HID * HID) {
    int which = t2 / (HID * HID), o = t2 % (HID * HID);
    const float* src = (which == 0) ? W2 : (which == 1) ? W3 : W4;
    W234h[t2] = (_Float16)src[o];
  }
}

// ---------------------------------------------------------------------------
// Kernel 2: argmax_j (|p_j|^2 - 2 g.p_j) per grid cell (|g|^2 term constant).
// ---------------------------------------------------------------------------
__global__ __launch_bounds__(256)
void argmax_kernel(const float* __restrict__ pts, int* __restrict__ idxOut) {
  __shared__ float spx[256];
  __shared__ float spy[256];
  int gid = blockIdx.x * 256 + threadIdx.x;
  int b = gid / NCELL;
  int g = gid % NCELL;
  int ci = g / ROWS, ri = g % ROWS;
  float gx = 70.0f * (float)ci / 127.0f;
  float gy = -35.0f + 70.0f * (float)ri / 127.0f;
  const float* P = pts + (size_t)b * NPTS * 3;
  float best = -3.4e38f;
  int bi = 0;
  for (int t0 = 0; t0 < NPTS; t0 += 256) {
    int j = t0 + threadIdx.x;
    spx[threadIdx.x] = P[j * 3 + 0];
    spy[threadIdx.x] = P[j * 3 + 1];
    __syncthreads();
    for (int j2 = 0; j2 < 256; ++j2) {
      float px = spx[j2], py = spy[j2];
      float s = px * px + py * py - 2.0f * (gx * px + gy * py);
      if (s > best) { best = s; bi = t0 + j2; }
    }
    __syncthreads();
  }
  idxOut[gid] = bi;
}

// ---------------------------------------------------------------------------
// Kernel 3: gather point + bilinear sample 960 pyramid channels + [dx,dy,z],
// write f16 X0 row (K padded to 992). One 128-thread block per cell.
// ---------------------------------------------------------------------------
__global__ __launch_bounds__(128)
void feat_kernel(const float* __restrict__ i0, const float* __restrict__ i1,
                 const float* __restrict__ i2, const float* __restrict__ i3,
                 const float* __restrict__ pts, const float* __restrict__ uv,
                 const int* __restrict__ idx, _Float16* __restrict__ X0) {
  int bg = blockIdx.x;
  int b = bg / NCELL, g = bg % NCELL;
  int t = threadIdx.x;
  int id = idx[bg];
  const float* P = pts + ((size_t)b * NPTS + id) * 3;
  float u_img = uv[((size_t)b * NPTS + id) * 2 + 0];
  float v_img = uv[((size_t)b * NPTS + id) * 2 + 1];
  _Float16* xrow = X0 + (size_t)bg * K1P;

  for (int cc = t; cc < 960; cc += 128) {
    const float* img; int C, H, W; float ds; int c;
    if (cc < 64)       { img = i0; C = 64;  H = 96; W = 312; ds = 4.0f;  c = cc; }
    else if (cc < 192) { img = i1; C = 128; H = 48; W = 156; ds = 8.0f;  c = cc - 64; }
    else if (cc < 448) { img = i2; C = 256; H = 24; W = 78;  ds = 16.0f; c = cc - 192; }
    else               { img = i3; C = 512; H = 12; W = 39;  ds = 32.0f; c = cc - 448; }
    float u = u_img / ds, v = v_img / ds;
    int ul = (int)u, vl = (int)v;
    float du = u - (float)ul, dv = v - (float)vl;
    const float* base = img + (((size_t)b * C + c) * H + vl) * W + ul;
    float val = base[0]     * (1.0f - dv) * (1.0f - du)
              + base[W]     * dv          * (1.0f - du)
              + base[1]     * (1.0f - dv) * du
              + base[W + 1] * dv          * du;
    xrow[cc] = (_Float16)val;
  }
  if (t == 0) {
    int ci = g / ROWS, ri = g % ROWS;
    float gx = 70.0f * (float)ci / 127.0f;
    float gy = -35.0f + 70.0f * (float)ri / 127.0f;
    xrow[960] = (_Float16)(P[0] - gx);
    xrow[961] = (_Float16)(P[1] - gy);
    xrow[962] = (_Float16)P[2];
    for (int k = 963; k < K1P; ++k) xrow[k] = (_Float16)0.0f;
  }
}

// ---------------------------------------------------------------------------
// Kernel 4: fused MLP. One WG = 64 rows of one batch; 4 waves x 16 rows.
// W2/3/4 async-staged into LDS (row-padded 128->136 halfs, conflict-free
// fragment reads), overlapped with the layer-1 global-K WMMA sweep.
//
// 16-bit A-fragment lane layout (ISA 7.12.2): element e of the v16h at lane L
// holds K = (e&7) + 8*(L>>4) + 16*(e>>3)  -> two contiguous v8h (b128) loads.
// B fragment uses the same pattern with N = lane&15 (W stored [N][K]).
// ---------------------------------------------------------------------------
__global__ __launch_bounds__(128)
void mlp_kernel(const _Float16* __restrict__ X0,
                const _Float16* __restrict__ W1h,
                const _Float16* __restrict__ W234h,
                const float* __restrict__ wz,
                const float* __restrict__ b1, const float* __restrict__ b2,
                const float* __restrict__ b3, const float* __restrict__ b4,
                const float* __restrict__ W5, const float* __restrict__ b5,
                const float* __restrict__ bev, float* __restrict__ out) {
  __shared__ _Float16 bufA[4][16 * WPAD];      // per-wave 16x128 act tiles
  __shared__ _Float16 bufB[4][16 * WPAD];
  __shared__ _Float16 wlds[3 * HID * WPAD];    // W2/3/4, row-padded
  __shared__ float w5s[HID];

  int wave = threadIdx.x >> 5;
  int lane = threadIdx.x & 31;
  int ln = lane & 15;   // N (or M) index within a 16-tile
  int hi = lane >> 4;   // half-wave selector

  int blk = blockIdx.x;            // 512 blocks = 2 batches * 256 M-tiles
  int b = blk >> 8;
  int mWG = (blk & 255) * 64;
  int grow = mWG + wave * 16 + ln; // this lane's row (A frags / output)

  if (threadIdx.x < HID) w5s[threadIdx.x] = W5[threadIdx.x];
  float b5v = b5[0];

  // ---- async-stage W2/3/4 -> LDS (6144 x b128 chunks), ASYNCcnt-tracked.
  // Overlaps with the layer-1 WMMA sweep below, which reads only global X0/W1.
  for (int j = threadIdx.x; j < 3 * HID * (HID / 8); j += 128) {
    int row = j >> 4, seg = j & 15;
    unsigned dst = (unsigned)(uintptr_t)&wlds[row * WPAD + seg * 8];
    unsigned long long src =
        (unsigned long long)(uintptr_t)(W234h + (size_t)row * HID + seg * 8);
    asm volatile("global_load_async_to_lds_b128 %0, %1, off"
                 :: "v"(dst), "v"(src) : "memory");
  }

  // ---- layer 1: H1b = X0 @ W1^T + b1, K = 992 (31 x k32 steps) ----
  v8f accH1[8];
#pragma unroll
  for (int nb = 0; nb < 8; ++nb) {
    float bv = b1[nb * 16 + ln];
#pragma unroll
    for (int r = 0; r < 8; ++r) accH1[nb][r] = bv;
  }
  const _Float16* arow = X0 + ((size_t)b * NCELL + grow) * K1P;
  __builtin_prefetch(arow, 0, 1);  // global_prefetch_b8
  for (int ks = 0; ks < 31; ++ks) {
    int kb = ks * 32;
    FragU A;
    A.h[0] = *(const v8h*)(arow + kb + 8 * hi);
    A.h[1] = *(const v8h*)(arow + kb + 16 + 8 * hi);
#pragma unroll
    for (int nb = 0; nb < 8; ++nb) {
      const _Float16* brow = W1h + (size_t)(nb * 16 + ln) * K1P + kb;
      FragU Bf;
      Bf.h[0] = *(const v8h*)(brow + 8 * hi);
      Bf.h[1] = *(const v8h*)(brow + 16 + 8 * hi);
      accH1[nb] = __builtin_amdgcn_wmma_f32_16x16x32_f16(
          false, A.v, false, Bf.v, (short)0, accH1[nb], false, false);
    }
  }

  float wzl[8];
#pragma unroll
  for (int nb = 0; nb < 8; ++nb) wzl[nb] = wz[nb * 16 + ln];

  // weights must be resident in LDS before the channel loop
  asm volatile("s_wait_asynccnt 0x0" ::: "memory");
  __syncthreads();

  // ---- per z-channel: shift/relu -> 3 hidden GEMMs -> dot -> residual ----
  for (int c = 0; c < CHA; ++c) {
    float zvc = 3.2f * (float)c / 7.0f;
    _Float16* cur = bufA[wave];
    _Float16* nxt = bufB[wave];

    // h1 = relu(h1b - zv*wz): C/D layout (M = r+8*hi, N = nb*16+ln) -> LDS
#pragma unroll
    for (int nb = 0; nb < 8; ++nb) {
#pragma unroll
      for (int r = 0; r < 8; ++r) {
        float v = accH1[nb][r] - zvc * wzl[nb];
        v = v > 0.0f ? v : 0.0f;
        cur[(r + 8 * hi) * WPAD + nb * 16 + ln] = (_Float16)v;
      }
    }
    __syncthreads();

    // layers 2..4: 128x128 GEMM + bias + relu, K = 128 (4 x k32 steps),
    // B fragments now served from LDS (ds_load_b128, conflict-free).
    for (int L = 0; L < 3; ++L) {
      const _Float16* Wl = wlds + (size_t)L * HID * WPAD;
      const float* bl = (L == 0) ? b2 : (L == 1) ? b3 : b4;
      v8f acc[8];
#pragma unroll
      for (int nb = 0; nb < 8; ++nb) {
        float bv = bl[nb * 16 + ln];
#pragma unroll
        for (int r = 0; r < 8; ++r) acc[nb][r] = bv;
      }
#pragma unroll
      for (int ks = 0; ks < 4; ++ks) {
        int kb = ks * 32;
        FragU A;
        A.h[0] = *(const v8h*)(cur + ln * WPAD + kb + 8 * hi);
        A.h[1] = *(const v8h*)(cur + ln * WPAD + kb + 16 + 8 * hi);
#pragma unroll
        for (int nb = 0; nb < 8; ++nb) {
          const _Float16* brow = Wl + (size_t)(nb * 16 + ln) * WPAD + kb;
          FragU Bf;
          Bf.h[0] = *(const v8h*)(brow + 8 * hi);
          Bf.h[1] = *(const v8h*)(brow + 16 + 8 * hi);
          acc[nb] = __builtin_amdgcn_wmma_f32_16x16x32_f16(
              false, A.v, false, Bf.v, (short)0, acc[nb], false, false);
        }
      }
#pragma unroll
      for (int nb = 0; nb < 8; ++nb) {
#pragma unroll
        for (int r = 0; r < 8; ++r) {
          float v = acc[nb][r];
          v = v > 0.0f ? v : 0.0f;
          nxt[(r + 8 * hi) * WPAD + nb * 16 + ln] = (_Float16)v;
        }
      }
      __syncthreads();
      _Float16* tmp = cur; cur = nxt; nxt = tmp;
    }

    // layer 5: o = relu(h4 . w5 + b5); each half-wave sums 64 K, shfl-combine
    float part = 0.0f;
    for (int k = 0; k < 64; ++k) {
      int kk = hi * 64 + k;
      part += (float)cur[ln * WPAD + kk] * w5s[kk];
    }
    float tot = part + __shfl_xor(part, 16, 32);
    if (hi == 0) {
      float o = tot + b5v;
      o = o > 0.0f ? o : 0.0f;
      size_t oidx = ((size_t)(b * CHA + c)) * NCELL + grow;
      out[oidx] = bev[oidx] + o;
    }
    __syncthreads();
  }
}

// ---------------------------------------------------------------------------
extern "C" void kernel_launch(void* const* d_in, const int* in_sizes, int n_in,
                              void* d_out, int out_size, void* d_ws, size_t ws_size,
                              hipStream_t stream) {
  const float* bev = (const float*)d_in[0];
  const float* i0  = (const float*)d_in[1];
  const float* i1  = (const float*)d_in[2];
  const float* i2  = (const float*)d_in[3];
  const float* i3  = (const float*)d_in[4];
  const float* pts = (const float*)d_in[5];
  const float* uv  = (const float*)d_in[6];
  // d_in[7] = num_points (int64), unused: always N=4096
  const float* W1 = (const float*)d_in[8];
  const float* b1 = (const float*)d_in[9];
  const float* W2 = (const float*)d_in[10];
  const float* b2 = (const float*)d_in[11];
  const float* W3 = (const float*)d_in[12];
  const float* b3 = (const float*)d_in[13];
  const float* W4 = (const float*)d_in[14];
  const float* b4 = (const float*)d_in[15];
  const float* W5 = (const float*)d_in[16];
  const float* b5 = (const float*)d_in[17];

  char* ws = (char*)d_ws;
  size_t off = 0;
  _Float16* X0 = (_Float16*)(ws + off); off += (size_t)2 * NCELL * K1P * sizeof(_Float16);
  int* IDX = (int*)(ws + off);          off += (size_t)2 * NCELL * sizeof(int);
  _Float16* W1h = (_Float16*)(ws + off);   off += (size_t)128 * K1P * sizeof(_Float16);
  _Float16* W234h = (_Float16*)(ws + off); off += (size_t)3 * HID * HID * sizeof(_Float16);
  float* wzb = (float*)(ws + off);         off += 128 * sizeof(float);
  (void)ws_size; (void)in_sizes; (void)n_in; (void)out_size;

  hipLaunchKernelGGL(wconv_kernel, dim3(688), dim3(256), 0, stream,
                     W1, W2, W3, W4, W1h, W234h, wzb);
  hipLaunchKernelGGL(argmax_kernel, dim3(128), dim3(256), 0, stream, pts, IDX);
  hipLaunchKernelGGL(feat_kernel, dim3(2 * NCELL), dim3(128), 0, stream,
                     i0, i1, i2, i3, pts, uv, IDX, X0);
  hipLaunchKernelGGL(mlp_kernel, dim3(512), dim3(128), 0, stream,
                     X0, W1h, W234h, wzb, b1, b2, b3, b4, W5, b5, bev, (float*)d_out);
}
